// NaViT_2078764172102
// MI455X (gfx1250) — matrix operand order repaired and checked by reading, not compiled, not run on hardware
//
#include <hip/hip_runtime.h>
#include <hip/hip_bf16.h>
#include <cstdint>
#include <cstddef>

typedef __attribute__((ext_vector_type(16))) __bf16 v16bf;
typedef __attribute__((ext_vector_type(8)))  float  v8f;

#define LMAXT 1024
#define NB    8
#define DIMD  512
#define NHEAD 8
#define DHD   64
#define NDEPTH 6
#define MLPD  2048
#define NCLS  1000
#define PDIM  768

__device__ const int g_L[8] = {1024, 768, 896, 576, 560, 512, 384, 640};
__device__ const int g_W16[8] = {32, 24, 32, 24, 28, 32, 16, 20};

// ---------------------------------------------------------------------------
// gfx1250 async global->LDS copy (16B), tracked by ASYNCcnt.
// VDST = LDS byte address (low 32 bits of generic LDS pointer), VADDR = 64-bit
// global address, GV mode (saddr = off).
// ---------------------------------------------------------------------------
__device__ __forceinline__ void async_ld_b128(void* lds, const void* g) {
  const uint32_t l = (uint32_t)(uintptr_t)lds;
  const uint64_t ga = (uint64_t)(uintptr_t)g;
  asm volatile("global_load_async_to_lds_b128 %0, %1, off" ::"v"(l), "v"(ga)
               : "memory");
}
#define S_WAIT_ASYNC_0() asm volatile("s_wait_asynccnt 0x0" ::: "memory")
#define S_WAIT_ASYNC_4() asm volatile("s_wait_asynccnt 0x4" ::: "memory")

// ---------------------------------------------------------------------------
// Pack images -> tokens [B, LMAX, 768]; padded rows zeroed.
// ---------------------------------------------------------------------------
__global__ __launch_bounds__(256) void k_pack(const float* __restrict__ images,
                                              float* __restrict__ tok) {
  const int b = blockIdx.y, l = blockIdx.x, tid = threadIdx.x;
  const int w = g_W16[b];
  const int L = g_L[b];
  float* dst = tok + ((size_t)b * LMAXT + l) * PDIM;
  if (l >= L) {
    for (int i = tid; i < PDIM; i += 256) dst[i] = 0.f;
    return;
  }
  const int hi = l / w, wi = l % w;
  for (int i = tid; i < PDIM; i += 256) {
    const int c = i >> 8, py = (i >> 4) & 15, px = i & 15;
    const int row = hi * 16 + py, col = wi * 16 + px;
    dst[i] = images[((size_t)b * 3 + c) * 512 * 512 + (size_t)row * 512 + col];
  }
}

// ---------------------------------------------------------------------------
// Row LayerNorm: one wave per row; optional f32 / bf16 outputs.
// ---------------------------------------------------------------------------
__global__ __launch_bounds__(32) void k_layernorm(const float* __restrict__ x,
                                                  const float* __restrict__ g,
                                                  const float* __restrict__ bias,
                                                  float* __restrict__ yf,
                                                  __bf16* __restrict__ ybf,
                                                  int N, float scale) {
  const size_t row = blockIdx.x;
  const float* xr = x + row * N;
  float s = 0.f, ss = 0.f;
  for (int i = threadIdx.x; i < N; i += 32) {
    const float v = xr[i];
    s += v; ss += v * v;
  }
  for (int off = 16; off; off >>= 1) {
    s  += __shfl_xor(s,  off, 32);
    ss += __shfl_xor(ss, off, 32);
  }
  const float mu = s / N;
  const float var = ss / N - mu * mu;
  const float rstd = rsqrtf(var + 1e-5f);
  for (int i = threadIdx.x; i < N; i += 32) {
    float v = (xr[i] - mu) * rstd * g[i] * scale;
    if (bias) v += bias[i];
    if (yf)  yf[row * N + i] = v;
    if (ybf) ybf[row * N + i] = (__bf16)v;
  }
}

__global__ __launch_bounds__(256) void k_cvt_bf16(const float* __restrict__ x,
                                                  __bf16* __restrict__ y, int n) {
  const int i = blockIdx.x * 256 + threadIdx.x;
  if (i < n) y[i] = (__bf16)x[i];
}

__global__ __launch_bounds__(256) void k_posmask(const float* __restrict__ y,
                                                 const float* __restrict__ pos_h,
                                                 const float* __restrict__ pos_w,
                                                 float* __restrict__ t) {
  const int b = blockIdx.y, l = blockIdx.x, tid = threadIdx.x;
  const int w = g_W16[b];
  const int L = g_L[b];
  float* dst = t + ((size_t)b * LMAXT + l) * DIMD;
  if (l >= L) {
    for (int i = tid; i < DIMD; i += 256) dst[i] = 0.f;
    return;
  }
  const int hi = l / w, wi = l % w;
  const float* src = y + ((size_t)b * LMAXT + l) * DIMD;
  for (int i = tid; i < DIMD; i += 256)
    dst[i] = src[i] + pos_h[hi * DIMD + i] + pos_w[wi * DIMD + i];
}

// ---------------------------------------------------------------------------
// FAST WMMA GEMM (full tiles only): C[M,N] = A[M,K] @ W[N,K]^T
// 256 threads = 8 waves; macro-tile 128x128x32; wave tile 64x32 (4x2 WMMA).
// Double-buffered LDS fed by async global->LDS copies (ASYNCcnt).
// Requires: M%128==0, N%128==0, K%32==0.
// ---------------------------------------------------------------------------
#define FTM 128
#define FTN 128
#define FTK 32

__global__ __launch_bounds__(256) void k_gemm_bf16_fast(
    const __bf16* __restrict__ A, const __bf16* __restrict__ W,
    const float* __restrict__ bias, const float* __restrict__ resid,
    float* __restrict__ Cf, __bf16* __restrict__ Cbf,
    int M, int N, int K, int fuse_gelu) {
  __shared__ __attribute__((aligned(16))) __bf16 sA[2][FTM * FTK];
  __shared__ __attribute__((aligned(16))) __bf16 sB[2][FTN * FTK];

  const int tid = threadIdx.x;
  const int lane = tid & 31;
  const int wave = tid >> 5;  // 0..7
  const int wr = wave >> 2;   // 0..1
  const int wc = wave & 3;    // 0..3
  const int m0 = blockIdx.y * FTM;
  const int n0 = blockIdx.x * FTN;

  const int lr0 = tid >> 2;          // rows 0..63 (chunk j adds 64)
  const int lc = (tid & 3) * 8;      // k-offset in elements

  auto issue = [&](int buf, int k0) {
    const __bf16* Ag = A + (size_t)m0 * K + k0;
    const __bf16* Wg = W + (size_t)n0 * K + k0;
#pragma unroll
    for (int j = 0; j < 2; ++j) {
      const int r = lr0 + j * 64;
      async_ld_b128(&sA[buf][r * FTK + lc], Ag + (size_t)r * K + lc);
    }
#pragma unroll
    for (int j = 0; j < 2; ++j) {
      const int r = lr0 + j * 64;
      async_ld_b128(&sB[buf][r * FTK + lc], Wg + (size_t)r * K + lc);
    }
  };

  v8f acc[4][2] = {};
  const int arow = lane & 15;
  const int akoff = (lane >> 4) * 16;
  const int bcol = lane & 15;
  const int bkhalf = (lane >> 4) * 16;

  issue(0, 0);
  const int nk = K / FTK;
  for (int kt = 0; kt < nk; ++kt) {
    const int buf = kt & 1;
    if (kt + 1 < nk) {
      issue(buf ^ 1, (kt + 1) * FTK);
      S_WAIT_ASYNC_4();   // 4 outstanding (next tile); current tile landed
    } else {
      S_WAIT_ASYNC_0();
    }
    __syncthreads();

    v16bf af[4], bf[2];
#pragma unroll
    for (int si = 0; si < 4; ++si) {
      const __bf16* pa = &sA[buf][(wr * 64 + si * 16 + arow) * FTK];
#pragma unroll
      for (int j = 0; j < 8; ++j) {
        af[si][j] = pa[akoff + j];
        af[si][8 + j] = pa[8 + akoff + j];
      }
    }
#pragma unroll
    for (int sj = 0; sj < 2; ++sj) {
      const __bf16* pb = &sB[buf][(wc * 32 + sj * 16 + bcol) * FTK + bkhalf];
#pragma unroll
      for (int j = 0; j < 16; ++j) bf[sj][j] = pb[j];
    }
#pragma unroll
    for (int si = 0; si < 4; ++si)
#pragma unroll
      for (int sj = 0; sj < 2; ++sj)
        acc[si][sj] = __builtin_amdgcn_wmma_f32_16x16x32_bf16(
            false, af[si], false, bf[sj], (short)0, acc[si][sj], false, false);
    __syncthreads();
  }

  const int ccol = lane & 15;
  const int crow0 = (lane >> 4) * 8;
#pragma unroll
  for (int si = 0; si < 4; ++si)
#pragma unroll
    for (int sj = 0; sj < 2; ++sj) {
      const int gcol = n0 + wc * 32 + sj * 16 + ccol;
#pragma unroll
      for (int r = 0; r < 8; ++r) {
        const int grow = m0 + wr * 64 + si * 16 + crow0 + r;
        float v = acc[si][sj][r];
        if (bias) v += bias[gcol];
        if (fuse_gelu) v = 0.5f * v * (1.0f + erff(v * 0.70710678f));
        const size_t idx = (size_t)grow * N + gcol;
        if (resid) v += resid[idx];
        if (Cf) Cf[idx] = v;
        if (Cbf) Cbf[idx] = (__bf16)v;
      }
    }
}

// ---------------------------------------------------------------------------
// Guarded WMMA GEMM for edge shapes (M=1/8, N=1000): tile 128x64x32.
// ---------------------------------------------------------------------------
#define TM 128
#define TN 64
#define TK 32

__global__ __launch_bounds__(256) void k_gemm_bf16(
    const __bf16* __restrict__ A, const __bf16* __restrict__ W,
    const float* __restrict__ bias, const float* __restrict__ resid,
    float* __restrict__ Cf, __bf16* __restrict__ Cbf,
    int M, int N, int K, int fuse_gelu) {
  __shared__ __attribute__((aligned(16))) __bf16 sA[TM * TK];
  __shared__ __attribute__((aligned(16))) __bf16 sB[TN * TK];

  const int tid = threadIdx.x;
  const int lane = tid & 31;
  const int wave = tid >> 5;
  const int wr = wave >> 1;
  const int wc = wave & 1;
  const int m0 = blockIdx.y * TM;
  const int n0 = blockIdx.x * TN;

  v8f acc[2][2] = {};
  const int arow = lane & 15;
  const int akoff = (lane >> 4) * 16;
  const int bcol = lane & 15;
  const int bkhalf = (lane >> 4) * 16;

  for (int k0 = 0; k0 < K; k0 += TK) {
    for (int i = tid; i < (TM * TK) / 8; i += 256) {
      const int r = i >> 2, c = (i & 3) * 8;
      const int gm = m0 + r;
      if (gm < M) {
        *(uint4*)&sA[r * TK + c] = *(const uint4*)(A + (size_t)gm * K + k0 + c);
      } else {
        uint4 z = {0u, 0u, 0u, 0u};
        *(uint4*)&sA[r * TK + c] = z;
      }
    }
    for (int i = tid; i < (TN * TK) / 8; i += 256) {
      const int r = i >> 2, c = (i & 3) * 8;
      const int gn = n0 + r;
      if (gn < N) {
        *(uint4*)&sB[r * TK + c] = *(const uint4*)(W + (size_t)gn * K + k0 + c);
      } else {
        uint4 z = {0u, 0u, 0u, 0u};
        *(uint4*)&sB[r * TK + c] = z;
      }
    }
    __syncthreads();

    v16bf afrag[2], bfrag[2];
#pragma unroll
    for (int s = 0; s < 2; ++s) {
      const __bf16* pa = &sA[(wr * 32 + s * 16 + arow) * TK];
#pragma unroll
      for (int j = 0; j < 8; ++j) {
        afrag[s][j] = pa[akoff + j];
        afrag[s][8 + j] = pa[8 + akoff + j];
      }
      const __bf16* pb = &sB[(wc * 32 + s * 16 + bcol) * TK + bkhalf];
#pragma unroll
      for (int j = 0; j < 16; ++j) bfrag[s][j] = pb[j];
    }
#pragma unroll
    for (int si = 0; si < 2; ++si)
#pragma unroll
      for (int sj = 0; sj < 2; ++sj)
        acc[si][sj] = __builtin_amdgcn_wmma_f32_16x16x32_bf16(
            false, afrag[si], false, bfrag[sj], (short)0, acc[si][sj], false, false);
    __syncthreads();
  }

  const int ccol = lane & 15;
  const int crow0 = (lane >> 4) * 8;
#pragma unroll
  for (int si = 0; si < 2; ++si)
#pragma unroll
    for (int sj = 0; sj < 2; ++sj) {
      const int gcol = n0 + wc * 32 + sj * 16 + ccol;
      if (gcol >= N) continue;
#pragma unroll
      for (int r = 0; r < 8; ++r) {
        const int grow = m0 + wr * 32 + si * 16 + crow0 + r;
        if (grow >= M) continue;
        float v = acc[si][sj][r];
        if (bias) v += bias[gcol];
        if (fuse_gelu) v = 0.5f * v * (1.0f + erff(v * 0.70710678f));
        const size_t idx = (size_t)grow * N + gcol;
        if (resid) v += resid[idx];
        if (Cf) Cf[idx] = v;
        if (Cbf) Cbf[idx] = (__bf16)v;
      }
    }
}

// ---------------------------------------------------------------------------
// Flash attention per (b*H+h, q-tile of 64): 128 threads = 4 waves, each wave
// owns 16 q rows. Q/K/V [B,L,H,DH] bf16; Q pre-scaled by 1/8.
// Q/K tiles arrive via async global->LDS; V staged through VGPRs (transpose).
// ---------------------------------------------------------------------------
__global__ __launch_bounds__(128) void k_flash_attn(const __bf16* __restrict__ Q,
                                                    const __bf16* __restrict__ Kk,
                                                    const __bf16* __restrict__ V,
                                                    __bf16* __restrict__ O) {
  __shared__ __attribute__((aligned(16))) __bf16 sQ[64 * 64];
  __shared__ __attribute__((aligned(16))) __bf16 sK[64 * 64];
  __shared__ __attribute__((aligned(16))) __bf16 sVt[64 * 64];  // [dh][key]
  __shared__ __attribute__((aligned(16))) __bf16 sP[4][16 * 64];

  const int bh = blockIdx.x;
  const int b = bh >> 3, h = bh & 7;
  const int q0 = blockIdx.y * 64;
  const int Lb = g_L[b];
  const int tid = threadIdx.x;
  const int lane = tid & 31;
  const int wave = tid >> 5;

  // async Q tile: 512 chunks of 16B / 128 threads = 4 per thread
#pragma unroll
  for (int j = 0; j < 4; ++j) {
    const int i = tid + j * 128;
    const int r = i >> 3, c = (i & 7) * 8;
    async_ld_b128(&sQ[r * 64 + c],
                  Q + (((size_t)b * LMAXT + q0 + r) * NHEAD + h) * DHD + c);
  }
  S_WAIT_ASYNC_0();
  __syncthreads();

  const int arow = lane & 15;
  const int akoff = (lane >> 4) * 16;
  const int bcol = lane & 15;
  const int bkhalf = (lane >> 4) * 16;
  const int ccol = lane & 15;
  const int crow0 = (lane >> 4) * 8;

  v16bf qf[2];
#pragma unroll
  for (int ks = 0; ks < 2; ++ks) {
    const __bf16* p = &sQ[(wave * 16 + arow) * 64 + ks * 32];
#pragma unroll
    for (int j = 0; j < 8; ++j) { qf[ks][j] = p[akoff + j]; qf[ks][8 + j] = p[8 + akoff + j]; }
  }

  v8f oacc[4] = {};
  float m8[8], l8[8];
#pragma unroll
  for (int r = 0; r < 8; ++r) { m8[r] = -1e30f; l8[r] = 0.f; }

  const int nkb = (Lb + 63) >> 6;
  for (int kb = 0; kb < nkb; ++kb) {
    __syncthreads();
    // async K tile
#pragma unroll
    for (int j = 0; j < 4; ++j) {
      const int i = tid + j * 128;
      const int r = i >> 3, c = (i & 7) * 8;
      async_ld_b128(&sK[r * 64 + c],
                    Kk + (((size_t)b * LMAXT + kb * 64 + r) * NHEAD + h) * DHD + c);
    }
    // V tile: batch 4 global loads, then LDS transpose scatter
    uint4 vv[4];
    int vr[4], vc[4];
#pragma unroll
    for (int j = 0; j < 4; ++j) {
      const int i = tid + j * 128;
      vr[j] = i >> 3; vc[j] = (i & 7) * 8;
      vv[j] = *(const uint4*)(V + (((size_t)b * LMAXT + kb * 64 + vr[j]) * NHEAD + h) * DHD + vc[j]);
    }
#pragma unroll
    for (int j = 0; j < 4; ++j) {
      const __bf16* e = (const __bf16*)&vv[j];
#pragma unroll
      for (int jj = 0; jj < 8; ++jj) sVt[(vc[j] + jj) * 64 + vr[j]] = e[jj];
    }
    S_WAIT_ASYNC_0();
    __syncthreads();

    // S = Q @ K^T : 16 rows x 64 keys per wave
    v8f s[4] = {};
#pragma unroll
    for (int ct = 0; ct < 4; ++ct)
#pragma unroll
      for (int ks = 0; ks < 2; ++ks) {
        v16bf bf;
        const __bf16* p = &sK[(ct * 16 + bcol) * 64 + ks * 32 + bkhalf];
#pragma unroll
        for (int j = 0; j < 16; ++j) bf[j] = p[j];
        s[ct] = __builtin_amdgcn_wmma_f32_16x16x32_bf16(false, qf[ks], false, bf,
                                                        (short)0, s[ct], false, false);
      }

    // key mask + online softmax
    float mnew[8];
#pragma unroll
    for (int r = 0; r < 8; ++r) mnew[r] = m8[r];
#pragma unroll
    for (int ct = 0; ct < 4; ++ct) {
      const int kg = kb * 64 + ct * 16 + ccol;
      const bool masked = (kg >= Lb);
#pragma unroll
      for (int r = 0; r < 8; ++r) {
        float sv = s[ct][r];
        if (masked) sv = -1e30f;
        s[ct][r] = sv;
        mnew[r] = fmaxf(mnew[r], sv);
      }
    }
#pragma unroll
    for (int off = 1; off < 16; off <<= 1)
#pragma unroll
      for (int r = 0; r < 8; ++r) mnew[r] = fmaxf(mnew[r], __shfl_xor(mnew[r], off, 32));

    float rs[8];
#pragma unroll
    for (int r = 0; r < 8; ++r) rs[r] = 0.f;
#pragma unroll
    for (int ct = 0; ct < 4; ++ct)
#pragma unroll
      for (int r = 0; r < 8; ++r) {
        const float p = __expf(s[ct][r] - mnew[r]);
        s[ct][r] = p;
        rs[r] += p;
      }
#pragma unroll
    for (int off = 1; off < 16; off <<= 1)
#pragma unroll
      for (int r = 0; r < 8; ++r) rs[r] += __shfl_xor(rs[r], off, 32);

#pragma unroll
    for (int r = 0; r < 8; ++r) {
      const float alpha = __expf(m8[r] - mnew[r]);
      l8[r] = l8[r] * alpha + rs[r];
      m8[r] = mnew[r];
#pragma unroll
      for (int ot = 0; ot < 4; ++ot) oacc[ot][r] *= alpha;
    }

    // stage P (C layout -> row-major) into this wave's private LDS region
    __bf16* pw = &sP[wave][0];
#pragma unroll
    for (int ct = 0; ct < 4; ++ct)
#pragma unroll
      for (int r = 0; r < 8; ++r)
        pw[(crow0 + r) * 64 + ct * 16 + ccol] = (__bf16)s[ct][r];

    // O += P @ V
#pragma unroll
    for (int ks = 0; ks < 2; ++ks) {
      v16bf pf;
      const __bf16* pp = &pw[arow * 64 + ks * 32];
#pragma unroll
      for (int j = 0; j < 8; ++j) { pf[j] = pp[akoff + j]; pf[8 + j] = pp[8 + akoff + j]; }
#pragma unroll
      for (int ot = 0; ot < 4; ++ot) {
        v16bf bf;
        const __bf16* p = &sVt[(ot * 16 + bcol) * 64 + ks * 32 + bkhalf];
#pragma unroll
        for (int j = 0; j < 16; ++j) bf[j] = p[j];
        oacc[ot] = __builtin_amdgcn_wmma_f32_16x16x32_bf16(false, pf, false, bf,
                                                           (short)0, oacc[ot], false, false);
      }
    }
  }

#pragma unroll
  for (int r = 0; r < 8; ++r) {
    const float inv = 1.0f / l8[r];
    const int qrow = q0 + wave * 16 + crow0 + r;
#pragma unroll
    for (int ot = 0; ot < 4; ++ot)
      O[(((size_t)b * LMAXT + qrow) * NHEAD + h) * DHD + ot * 16 + ccol] =
          (__bf16)(oacc[ot][r] * inv);
  }
}

// ---------------------------------------------------------------------------
// Pooling attention: 1 query per (b,h); 256 threads per block.
// ---------------------------------------------------------------------------
__global__ __launch_bounds__(256) void k_pool_attn(const float* __restrict__ qh,
                                                   const __bf16* __restrict__ Kp,
                                                   const __bf16* __restrict__ Vp,
                                                   float* __restrict__ out) {
  __shared__ float sq[DHD];
  __shared__ float red[256];
  __shared__ float sp[LMAXT];
  const int b = blockIdx.x, h = blockIdx.y, tid = threadIdx.x;
  const int Lb = g_L[b];
  if (tid < DHD) sq[tid] = qh[h * DHD + tid];
  __syncthreads();
  float sv[4], mx = -1e30f;
  for (int it = 0; it < 4; ++it) {
    const int kg = tid + it * 256;
    float s = -1e30f;
    if (kg < Lb) {
      const __bf16* kr = Kp + (((size_t)b * LMAXT + kg) * NHEAD + h) * DHD;
      float acc = 0.f;
      for (int d = 0; d < DHD; ++d) acc += sq[d] * (float)kr[d];
      s = acc;
    }
    sv[it] = s;
    mx = fmaxf(mx, s);
  }
  red[tid] = mx; __syncthreads();
  for (int o = 128; o; o >>= 1) { if (tid < o) red[tid] = fmaxf(red[tid], red[tid + o]); __syncthreads(); }
  mx = red[0]; __syncthreads();
  float ls = 0.f;
  for (int it = 0; it < 4; ++it) {
    const float p = (sv[it] > -1e29f) ? __expf(sv[it] - mx) : 0.f;
    sp[tid + it * 256] = p;
    ls += p;
  }
  red[tid] = ls; __syncthreads();
  for (int o = 128; o; o >>= 1) { if (tid < o) red[tid] += red[tid + o]; __syncthreads(); }
  const float inv = 1.0f / red[0];
  __syncthreads();
  if (tid < DHD) {
    float acc = 0.f;
    for (int k = 0; k < Lb; ++k)
      acc += sp[k] * (float)Vp[(((size_t)b * LMAXT + k) * NHEAD + h) * DHD + tid];
    out[(size_t)b * (NHEAD * DHD) + h * DHD + tid] = acc * inv;
  }
}

// ---------------------------------------------------------------------------
extern "C" void kernel_launch(void* const* d_in, const int* in_sizes, int n_in,
                              void* d_out, int out_size, void* d_ws, size_t ws_size,
                              hipStream_t stream) {
  (void)in_sizes; (void)n_in; (void)out_size; (void)ws_size;
  const float* images   = (const float*)d_in[0];
  const float* pe_ln1_w = (const float*)d_in[1];
  const float* pe_ln1_b = (const float*)d_in[2];
  const float* pe_w     = (const float*)d_in[3];
  const float* pe_b     = (const float*)d_in[4];
  const float* pe_ln2_w = (const float*)d_in[5];
  const float* pe_ln2_b = (const float*)d_in[6];
  const float* pos_h    = (const float*)d_in[7];
  const float* pos_w    = (const float*)d_in[8];
  const float* attn_ln  = (const float*)d_in[9];
  const float* wq       = (const float*)d_in[10];
  const float* wk       = (const float*)d_in[11];
  const float* wv       = (const float*)d_in[12];
  const float* qn       = (const float*)d_in[13];
  const float* kn       = (const float*)d_in[14];
  const float* wo       = (const float*)d_in[15];
  const float* ff_ln    = (const float*)d_in[16];
  const float* ff_w1    = (const float*)d_in[17];
  const float* ff_b1    = (const float*)d_in[18];
  const float* ff_w2    = (const float*)d_in[19];
  const float* ff_b2    = (const float*)d_in[20];
  const float* final_ln = (const float*)d_in[21];
  const float* pool_q   = (const float*)d_in[22];
  const float* pool_ln  = (const float*)d_in[23];
  const float* pool_wq  = (const float*)d_in[24];
  const float* pool_wk  = (const float*)d_in[25];
  const float* pool_wv  = (const float*)d_in[26];
  const float* pool_qn  = (const float*)d_in[27];
  const float* pool_kn  = (const float*)d_in[28];
  const float* pool_wo  = (const float*)d_in[29];
  const float* head_ln  = (const float*)d_in[30];
  const float* head_w   = (const float*)d_in[31];

  const int ROWS = NB * LMAXT;  // 8192

  char* w = (char*)d_ws;
  size_t off = 0;
  auto alloc = [&](size_t bytes) -> void* {
    void* p = w + off;
    off = (off + bytes + 255) & ~(size_t)255;
    return p;
  };
  float*  tok    = (float*) alloc((size_t)ROWS * PDIM * 4);
  __bf16* tokbf  = (__bf16*)alloc((size_t)ROWS * PDIM * 2);
  float*  t      = (float*) alloc((size_t)ROWS * DIMD * 4);
  float*  qf     = (float*) alloc((size_t)ROWS * DIMD * 4);
  float*  kf     = (float*) alloc((size_t)ROWS * DIMD * 4);
  __bf16* xnbf   = (__bf16*)alloc((size_t)ROWS * DIMD * 2);
  __bf16* qbf    = (__bf16*)alloc((size_t)ROWS * DIMD * 2);
  __bf16* kbf    = (__bf16*)alloc((size_t)ROWS * DIMD * 2);
  __bf16* vbf    = (__bf16*)alloc((size_t)ROWS * DIMD * 2);
  __bf16* obf    = (__bf16*)alloc((size_t)ROWS * DIMD * 2);
  __bf16* h1bf   = (__bf16*)alloc((size_t)ROWS * MLPD * 2);
  __bf16* pewbf  = (__bf16*)alloc((size_t)DIMD * PDIM * 2);
  __bf16* wqbf   = (__bf16*)alloc((size_t)NDEPTH * DIMD * DIMD * 2);
  __bf16* wkbf   = (__bf16*)alloc((size_t)NDEPTH * DIMD * DIMD * 2);
  __bf16* wvbf   = (__bf16*)alloc((size_t)NDEPTH * DIMD * DIMD * 2);
  __bf16* wobf   = (__bf16*)alloc((size_t)NDEPTH * DIMD * DIMD * 2);
  __bf16* ff1bf  = (__bf16*)alloc((size_t)NDEPTH * MLPD * DIMD * 2);
  __bf16* ff2bf  = (__bf16*)alloc((size_t)NDEPTH * DIMD * MLPD * 2);
  __bf16* pwqbf  = (__bf16*)alloc((size_t)DIMD * DIMD * 2);
  __bf16* pwkbf  = (__bf16*)alloc((size_t)DIMD * DIMD * 2);
  __bf16* pwvbf  = (__bf16*)alloc((size_t)DIMD * DIMD * 2);
  __bf16* pwobf  = (__bf16*)alloc((size_t)DIMD * DIMD * 2);
  __bf16* hwbf   = (__bf16*)alloc((size_t)NCLS * DIMD * 2);
  __bf16* qrowbf = (__bf16*)alloc(DIMD * 2);
  float*  qproj  = (float*) alloc(DIMD * 4);
  float*  qhf    = (float*) alloc(DIMD * 4);
  float*  poolo  = (float*) alloc((size_t)NB * DIMD * 4);
  __bf16* poolobf= (__bf16*)alloc((size_t)NB * DIMD * 2);
  float*  pooled = (float*) alloc((size_t)NB * DIMD * 4);
  __bf16* pooledbf=(__bf16*)alloc((size_t)NB * DIMD * 2);

  auto cvt = [&](const float* x, __bf16* y, int n) {
    k_cvt_bf16<<<(n + 255) / 256, 256, 0, stream>>>(x, y, n);
  };
  auto gemm = [&](const __bf16* A, const __bf16* Wt, const float* bias,
                  const float* resid, float* Cf, __bf16* Cbf,
                  int M, int N, int K, int gelu) {
    if ((M % FTM == 0) && (N % FTN == 0) && (K % FTK == 0)) {
      dim3 g(N / FTN, M / FTM);
      k_gemm_bf16_fast<<<g, 256, 0, stream>>>(A, Wt, bias, resid, Cf, Cbf, M, N, K, gelu);
    } else {
      dim3 g((N + TN - 1) / TN, (M + TM - 1) / TM);
      k_gemm_bf16<<<g, 256, 0, stream>>>(A, Wt, bias, resid, Cf, Cbf, M, N, K, gelu);
    }
  };
  auto ln = [&](const float* x, const float* g, const float* bias, float* yf,
                __bf16* ybf, int rows, int N, float scale) {
    k_layernorm<<<rows, 32, 0, stream>>>(x, g, bias, yf, ybf, N, scale);
  };

  // weight conversions (deterministic every call)
  cvt(pe_w, pewbf, DIMD * PDIM);
  cvt(wq, wqbf, NDEPTH * DIMD * DIMD);
  cvt(wk, wkbf, NDEPTH * DIMD * DIMD);
  cvt(wv, wvbf, NDEPTH * DIMD * DIMD);
  cvt(wo, wobf, NDEPTH * DIMD * DIMD);
  cvt(ff_w1, ff1bf, NDEPTH * MLPD * DIMD);
  cvt(ff_w2, ff2bf, NDEPTH * DIMD * MLPD);
  cvt(pool_wq, pwqbf, DIMD * DIMD);
  cvt(pool_wk, pwkbf, DIMD * DIMD);
  cvt(pool_wv, pwvbf, DIMD * DIMD);
  cvt(pool_wo, pwobf, DIMD * DIMD);
  cvt(head_w, hwbf, NCLS * DIMD);

  // patch embed
  k_pack<<<dim3(LMAXT, NB), 256, 0, stream>>>(images, tok);
  ln(tok, pe_ln1_w, pe_ln1_b, nullptr, tokbf, ROWS, PDIM, 1.f);
  gemm(tokbf, pewbf, pe_b, nullptr, qf, nullptr, ROWS, DIMD, PDIM, 0);
  ln(qf, pe_ln2_w, pe_ln2_b, kf, nullptr, ROWS, DIMD, 1.f);
  k_posmask<<<dim3(LMAXT, NB), 256, 0, stream>>>(kf, pos_h, pos_w, t);

  // transformer layers
  for (int l = 0; l < NDEPTH; ++l) {
    ln(t, attn_ln + (size_t)l * DIMD, nullptr, nullptr, xnbf, ROWS, DIMD, 1.f);
    gemm(xnbf, wqbf + (size_t)l * DIMD * DIMD, nullptr, nullptr, qf, nullptr, ROWS, DIMD, DIMD, 0);
    gemm(xnbf, wkbf + (size_t)l * DIMD * DIMD, nullptr, nullptr, kf, nullptr, ROWS, DIMD, DIMD, 0);
    gemm(xnbf, wvbf + (size_t)l * DIMD * DIMD, nullptr, nullptr, nullptr, vbf, ROWS, DIMD, DIMD, 0);
    ln(qf, qn + (size_t)l * DHD, nullptr, nullptr, qbf, ROWS * NHEAD, DHD, 0.125f);
    ln(kf, kn + (size_t)l * DHD, nullptr, nullptr, kbf, ROWS * NHEAD, DHD, 1.f);
    k_flash_attn<<<dim3(NB * NHEAD, LMAXT / 64), 128, 0, stream>>>(qbf, kbf, vbf, obf);
    gemm(obf, wobf + (size_t)l * DIMD * DIMD, nullptr, t, t, nullptr, ROWS, DIMD, DIMD, 0);
    ln(t, ff_ln + (size_t)l * DIMD, nullptr, nullptr, xnbf, ROWS, DIMD, 1.f);
    gemm(xnbf, ff1bf + (size_t)l * MLPD * DIMD, ff_b1 + (size_t)l * MLPD,
         nullptr, nullptr, h1bf, ROWS, MLPD, DIMD, 1);
    gemm(h1bf, ff2bf + (size_t)l * DIMD * MLPD, ff_b2 + (size_t)l * DIMD,
         t, t, nullptr, ROWS, DIMD, MLPD, 0);
  }

  // final norm + attention pooling + head
  ln(t, final_ln, nullptr, nullptr, xnbf, ROWS, DIMD, 1.f);
  ln(pool_q, pool_ln, nullptr, nullptr, qrowbf, 1, DIMD, 1.f);
  gemm(qrowbf, pwqbf, nullptr, nullptr, qproj, nullptr, 1, DIMD, DIMD, 0);
  ln(qproj, pool_qn, nullptr, qhf, nullptr, NHEAD, DHD, 0.125f);
  gemm(xnbf, pwkbf, nullptr, nullptr, kf, nullptr, ROWS, DIMD, DIMD, 0);
  ln(kf, pool_kn, nullptr, nullptr, kbf, ROWS * NHEAD, DHD, 1.f);
  gemm(xnbf, pwvbf, nullptr, nullptr, nullptr, vbf, ROWS, DIMD, DIMD, 0);
  k_pool_attn<<<dim3(NB, NHEAD), 256, 0, stream>>>(qhf, kbf, vbf, poolo);
  k_cvt_bf16<<<(NB * DIMD + 255) / 256, 256, 0, stream>>>(poolo, poolobf, NB * DIMD);
  gemm(poolobf, pwobf, nullptr, nullptr, pooled, nullptr, NB, DIMD, DIMD, 0);
  ln(pooled, head_ln, nullptr, nullptr, pooledbf, NB, DIMD, 1.f);
  gemm(pooledbf, hwbf, nullptr, nullptr, (float*)d_out, nullptr, NB, NCLS, DIMD, 0);
}
